// NodeNetwork_49349174231511
// MI455X (gfx1250) — matrix operand order repaired and checked by reading, not compiled, not run on hardware
//
#include <hip/hip_runtime.h>

// ---------------------------------------------------------------------------
// NodeNetwork fused inference for gfx1250 (MI455X), wave32 + WMMA bf16.
//
//   r1 = tanh( relu( [feat|hid] @ W1a + b1a ) @ W1b + b1b )      (cols 0..63)
//   r2 = tanh( relu( sum_d(mailbox) @ W2a + b2a ) @ W2b + b2b )  (cols 64..127)
//   out = [r1|r2] / ||[r1|r2]||_2 (per row)
//
// One wave handles a 16-row tile; 4 waves per block; 12500 tiles exactly.
// Weights are pre-transposed to bf16 (N-major) into d_ws by a prep kernel so
// B-fragments are contiguous 16B loads (stay resident in 192MB L2).
// ---------------------------------------------------------------------------

typedef __attribute__((ext_vector_type(16))) __bf16       v16bf;
typedef __attribute__((ext_vector_type(8)))  float        v8f;
typedef __attribute__((ext_vector_type(4)))  unsigned int v4u;

union FragU {
  v4u   q[2];
  v16bf bf;
};

// round-to-nearest-even (used only in one-shot weight prep)
__device__ __forceinline__ unsigned short f2bf_rne(float f) {
  unsigned int u = __builtin_bit_cast(unsigned int, f);
  u += 0x7FFFu + ((u >> 16) & 1u);
  return (unsigned short)(u >> 16);
}

// truncating bf16 (hot path; foldable to v_perm_b32)
__device__ __forceinline__ unsigned short f2bf_tr(float f) {
  return (unsigned short)(__builtin_bit_cast(unsigned int, f) >> 16);
}
__device__ __forceinline__ unsigned int pack_bf16(float lo, float hi) {
  const unsigned int ul = __builtin_bit_cast(unsigned int, lo);
  const unsigned int uh = __builtin_bit_cast(unsigned int, hi);
  return (uh & 0xFFFF0000u) | (ul >> 16);
}

// hardware transcendental tanh; v_nop covers the 1-op TRANS result hazard
__device__ __forceinline__ float htanh(float x) {
  float y;
  asm("v_tanh_f32 %0, %1\n\tv_nop" : "=v"(y) : "v"(x));
  return y;
}

// A-matrix 16x32 bf16 fragment (ISA 7.12.2): lane = 16*half + m,
// element j -> K = j + 8*((j>>3) + half)  => two contiguous 8-elem chunks.
__device__ __forceinline__ v16bf load_a_frag(const unsigned short* lds,
                                             int m, int stride, int k0, int half) {
  const int o0 = m * stride + k0 + 8 * half;
  FragU u;
  u.q[0] = *(const v4u*)(lds + o0);        // K = k0+8h .. +7
  u.q[1] = *(const v4u*)(lds + o0 + 16);   // K = k0+16+8h .. +7
  return u.bf;
}

// B-matrix 32x16 bf16 fragment: lane = 16*kh + n, element j -> K = j + 8*((j>>3) + kh).
// wT is N-major (row n holds K 0..Kdim-1 contiguously) => two contiguous 8-elem chunks.
__device__ __forceinline__ v16bf load_b_frag(const unsigned short* __restrict__ wT,
                                             int nrow, int Kdim, int k0, int kh) {
  const unsigned short* p = wT + nrow * Kdim + k0 + kh * 8;
  FragU u;
  u.q[0] = *(const v4u*)(p);
  u.q[1] = *(const v4u*)(p + 16);
  return u.bf;
}

// d_ws bf16 element offsets
#define W1AT_OFF 0       // [96][128]
#define W1BT_OFF 12288   // [64][96]
#define W2AT_OFF 18432   // [64][64]
#define W2BT_OFF 22528   // [64][64]
#define W_TOTAL  26624

__global__ __launch_bounds__(256)
void prep_weights(const float* __restrict__ w1a, const float* __restrict__ w1b,
                  const float* __restrict__ w2a, const float* __restrict__ w2b,
                  unsigned short* __restrict__ ws) {
  const int idx = blockIdx.x * 256 + threadIdx.x;
  if (idx >= W_TOTAL) return;
  float v;
  if (idx < W1BT_OFF) {                 // w1a: [128,96] -> [96][128]
    const int n = idx >> 7, k = idx & 127;
    v = w1a[k * 96 + n];
  } else if (idx < W2AT_OFF) {          // w1b: [96,64] -> [64][96]
    const int i = idx - W1BT_OFF, n = i / 96, k = i % 96;
    v = w1b[k * 64 + n];
  } else if (idx < W2BT_OFF) {          // w2a: [64,64] -> [64][64]
    const int i = idx - W2AT_OFF, n = i >> 6, k = i & 63;
    v = w2a[k * 64 + n];
  } else {                              // w2b: [64,64] -> [64][64]
    const int i = idx - W2BT_OFF, n = i >> 6, k = i & 63;
    v = w2b[k * 64 + n];
  }
  ws[idx] = f2bf_rne(v);
}

__global__ __launch_bounds__(128)
void node_net_fused(const float* __restrict__ feat,
                    const float* __restrict__ hid,
                    const float* __restrict__ mbox,
                    const unsigned short* __restrict__ wts,
                    const float* __restrict__ b1a,
                    const float* __restrict__ b1b,
                    const float* __restrict__ b2a,
                    const float* __restrict__ b2b,
                    float* __restrict__ out) {
  __shared__ __align__(16) unsigned short a1_lds[4][16 * 128];  // concat input, bf16
  __shared__ __align__(16) unsigned short x2_lds[4][16 * 64];   // mailbox sum, bf16
  __shared__ __align__(16) unsigned short h_lds [4][16 * 96];   // hidden acts, bf16

  const int wave = threadIdx.x >> 5;
  const int lane = threadIdx.x & 31;
  const int n    = lane & 15;    // N-column / A-row within tile
  const int kh   = lane >> 4;    // upper/lower half of wave
  const int base = (blockIdx.x * 4 + wave) * 16;   // 12500 tiles exactly

  unsigned short* myA1 = a1_lds[wave];
  unsigned short* myX2 = x2_lds[wave];
  unsigned short* myH  = h_lds[wave];

  // ---- stage concat(feat, hid) -> bf16 LDS (coalesced float4 loads) ----
  {
    const float* srcb = (kh == 0) ? feat : hid;
    const int c4 = n * 4;            // column in 64-wide source
    const int kb = kh * 64 + c4;     // column in 128-wide concat
    for (int r = 0; r < 16; ++r) {
      const float4 v = *(const float4*)(srcb + (size_t)(base + r) * 64 + c4);
      uint2 pk;
      pk.x = pack_bf16(v.x, v.y);
      pk.y = pack_bf16(v.z, v.w);
      *(uint2*)(myA1 + r * 128 + kb) = pk;
    }
  }

  // ---- mailbox sum-aggregate -> bf16 LDS ----
  // float4 per lane; half-waves split even/odd mailbox slots, combined with
  // one cross-half shuffle per component. 512B contiguous per load instr.
  for (int r = 0; r < 16; ++r) {
    const float* p = mbox + (size_t)(base + r) * 1024 + kh * 64 + n * 4;
    float4 a;
    a.x = 0.f; a.y = 0.f; a.z = 0.f; a.w = 0.f;
#pragma unroll
    for (int db = 0; db < 8; ++db) {          // d = 2*db + kh
      const float4 v = *(const float4*)(p + db * 128);
      a.x += v.x; a.y += v.y; a.z += v.z; a.w += v.w;
    }
    a.x += __shfl_xor(a.x, 16, 32);
    a.y += __shfl_xor(a.y, 16, 32);
    a.z += __shfl_xor(a.z, 16, 32);
    a.w += __shfl_xor(a.w, 16, 32);
    if (kh == 0) {
      uint2 pk;
      pk.x = pack_bf16(a.x, a.y);
      pk.y = pack_bf16(a.z, a.w);
      *(uint2*)(myX2 + r * 64 + n * 4) = pk;
    }
  }

  __syncthreads();

  // ================= Branch 1 =================
  // GEMM1a: [16,128] @ [128,96]  -> 6 N-tiles x 4 K-chunks = 24 WMMAs
  v8f acc1[6] = {};
#pragma unroll
  for (int kc = 0; kc < 4; ++kc) {
    const v16bf a = load_a_frag(myA1, n, 128, kc * 32, kh);
#pragma unroll
    for (int t = 0; t < 6; ++t) {
      const v16bf b = load_b_frag(wts + W1AT_OFF, t * 16 + n, 128, kc * 32, kh);
      acc1[t] = __builtin_amdgcn_wmma_f32_16x16x32_bf16(
          false, a, false, b, (short)0, acc1[t], false, false);
    }
  }
  // bias + ReLU, re-layout D(lane=N) -> A(lane=M) through LDS, stride 96
#pragma unroll
  for (int t = 0; t < 6; ++t) {
    const float bv = b1a[t * 16 + n];
#pragma unroll
    for (int r = 0; r < 8; ++r) {
      const float v = fmaxf(acc1[t][r] + bv, 0.f);
      myH[(r + 8 * kh) * 96 + t * 16 + n] = f2bf_tr(v);
    }
  }
  asm volatile("" ::: "memory");   // keep LDS store->load order (in-order per wave)

  // GEMM1b: [16,96] @ [96,64] -> 4 x 3 = 12 WMMAs
  v8f accB[4] = {};
#pragma unroll
  for (int kc = 0; kc < 3; ++kc) {
    const v16bf a = load_a_frag(myH, n, 96, kc * 32, kh);
#pragma unroll
    for (int t = 0; t < 4; ++t) {
      const v16bf b = load_b_frag(wts + W1BT_OFF, t * 16 + n, 96, kc * 32, kh);
      accB[t] = __builtin_amdgcn_wmma_f32_16x16x32_bf16(
          false, a, false, b, (short)0, accB[t], false, false);
    }
  }
  float res1[4][8];
#pragma unroll
  for (int t = 0; t < 4; ++t) {
    const float bv = b1b[t * 16 + n];
#pragma unroll
    for (int r = 0; r < 8; ++r) res1[t][r] = htanh(accB[t][r] + bv);
  }

  // ================= Branch 2 =================
  // GEMM2a: [16,64] @ [64,64] -> 4 x 2 = 8 WMMAs
  v8f accC[4] = {};
#pragma unroll
  for (int kc = 0; kc < 2; ++kc) {
    const v16bf a = load_a_frag(myX2, n, 64, kc * 32, kh);
#pragma unroll
    for (int t = 0; t < 4; ++t) {
      const v16bf b = load_b_frag(wts + W2AT_OFF, t * 16 + n, 64, kc * 32, kh);
      accC[t] = __builtin_amdgcn_wmma_f32_16x16x32_bf16(
          false, a, false, b, (short)0, accC[t], false, false);
    }
  }
  asm volatile("" ::: "memory");   // h1 fully consumed; reuse myH for h2
#pragma unroll
  for (int t = 0; t < 4; ++t) {
    const float bv = b2a[t * 16 + n];
#pragma unroll
    for (int r = 0; r < 8; ++r) {
      const float v = fmaxf(accC[t][r] + bv, 0.f);
      myH[(r + 8 * kh) * 64 + t * 16 + n] = f2bf_tr(v);
    }
  }
  asm volatile("" ::: "memory");

  // GEMM2b: [16,64] @ [64,64] -> 8 WMMAs
  v8f accD[4] = {};
#pragma unroll
  for (int kc = 0; kc < 2; ++kc) {
    const v16bf a = load_a_frag(myH, n, 64, kc * 32, kh);
#pragma unroll
    for (int t = 0; t < 4; ++t) {
      const v16bf b = load_b_frag(wts + W2BT_OFF, t * 16 + n, 64, kc * 32, kh);
      accD[t] = __builtin_amdgcn_wmma_f32_16x16x32_bf16(
          false, a, false, b, (short)0, accD[t], false, false);
    }
  }
  float res2[4][8];
#pragma unroll
  for (int t = 0; t < 4; ++t) {
    const float bv = b2b[t * 16 + n];
#pragma unroll
    for (int r = 0; r < 8; ++r) res2[t][r] = htanh(accD[t][r] + bv);
  }

  // ---- per-row L2 normalize (reduce over the 16-lane N group) + store ----
#pragma unroll
  for (int r = 0; r < 8; ++r) {
    float s = 0.f;
#pragma unroll
    for (int t = 0; t < 4; ++t)
      s += res1[t][r] * res1[t][r] + res2[t][r] * res2[t][r];
    s += __shfl_xor(s, 1, 32);
    s += __shfl_xor(s, 2, 32);
    s += __shfl_xor(s, 4, 32);
    s += __shfl_xor(s, 8, 32);       // masks < 16: stays within half-wave group
    const float inv = rsqrtf(s);
    float* op = out + (size_t)(base + r + 8 * kh) * 128;
#pragma unroll
    for (int t = 0; t < 4; ++t) {
      op[t * 16 + n]      = res1[t][r] * inv;   // r1 -> cols 0..63
      op[64 + t * 16 + n] = res2[t][r] * inv;   // r2 -> cols 64..127
    }
  }
}

extern "C" void kernel_launch(void* const* d_in, const int* in_sizes, int n_in,
                              void* d_out, int out_size, void* d_ws, size_t ws_size,
                              hipStream_t stream) {
  (void)in_sizes; (void)n_in; (void)out_size; (void)ws_size;
  const float* feat = (const float*)d_in[0];
  const float* hid  = (const float*)d_in[1];
  const float* mbox = (const float*)d_in[2];
  const float* w1a  = (const float*)d_in[3];
  const float* b1a  = (const float*)d_in[4];
  const float* w1b  = (const float*)d_in[5];
  const float* b1b  = (const float*)d_in[6];
  const float* w2a  = (const float*)d_in[7];
  const float* b2a  = (const float*)d_in[8];
  const float* w2b  = (const float*)d_in[9];
  const float* b2b  = (const float*)d_in[10];
  unsigned short* wts = (unsigned short*)d_ws;   // needs 53,248 B

  prep_weights<<<(W_TOTAL + 255) / 256, 256, 0, stream>>>(w1a, w1b, w2a, w2b, wts);
  // 200000 rows = 12500 tiles = 3125 blocks * 4 waves * 16 rows (exact)
  node_net_fused<<<3125, 128, 0, stream>>>(feat, hid, mbox, wts,
                                           b1a, b1b, b2a, b2b, (float*)d_out);
}
